// ENCODER_69114613730336
// MI455X (gfx1250) — matrix operand order repaired and checked by reading, not compiled, not run on hardware
//
#include <hip/hip_runtime.h>
#include <stdint.h>

#define L_  4
#define B_  2
#define S_  2048
#define D_  768
#define H_  12
#define F_  3072
#define HD_ 64
#define BS_ (B_*S_)
#define EPS_ 1e-5f

typedef __attribute__((ext_vector_type(16))) __bf16    bf16x16;
typedef __attribute__((ext_vector_type(8)))  float     f32x8;
typedef __attribute__((ext_vector_type(4)))  uint32_t  u32x4;

__device__ __forceinline__ uint16_t f2bf(float f) {
    union { float f; uint32_t u; } a; a.f = f;
    uint32_t u = a.u;
    u += 0x7fffu + ((u >> 16) & 1u);          // round-to-nearest-even
    return (uint16_t)(u >> 16);
}

// --------------------------------------------------------------------------
// CDNA5 async global->LDS path (ASYNCcnt), with sync fallback
// --------------------------------------------------------------------------
#if defined(__has_builtin)
# if __has_builtin(__builtin_amdgcn_global_load_async_to_lds_b128)
#  define HAVE_ASYNC_LDS 1
# endif
#endif

#ifdef HAVE_ASYNC_LDS
# if __has_builtin(__builtin_amdgcn_s_wait_asynccnt)
#  define WAIT_ASYNC(n) __builtin_amdgcn_s_wait_asynccnt(n)
# else
#  define WAIT_ASYNC(n) asm volatile("s_wait_asynccnt %0" :: "i"(n) : "memory")
# endif
// builtin signature: void(int4 AS1*, int4 AS3*, imm int, imm int)
typedef __attribute__((__vector_size__(4 * sizeof(int)))) int i32v4;
typedef __attribute__((address_space(1))) i32v4 gbl_i32v4;
typedef __attribute__((address_space(3))) i32v4 lds_i32v4;
__device__ __forceinline__ gbl_i32v4* to_gbl(const void* p) {
    return (gbl_i32v4*)(uintptr_t)p;            // generic global VA == AS1 VA
}
__device__ __forceinline__ lds_i32v4* to_lds(const void* p) {
    return (lds_i32v4*)(uint32_t)(uintptr_t)p;  // generic LDS addr[31:0] == LDS offset
}
#endif

// ---------------------------------------------------------------------------
// Generic NT GEMM:  C[M,N] = A[M,K] * B[N,K]^T (+bias) (ReLU?)  bf16 in, f32 acc
// block = 256 threads (8 waves, 2x4), tile 128x128, K-step 32
// ---------------------------------------------------------------------------
#define BM 128
#define BN 128
#define BK 32
#define LSTR 40   // padded LDS row stride in halves: 80B = 5x16B (b128-aligned)

__global__ __launch_bounds__(256) void k_gemm(
    const uint16_t* __restrict__ A,  int lda, long long sAz,
    const uint16_t* __restrict__ Bm, int ldb, long long sBz,
    const float* __restrict__ bias,
    float* __restrict__ Cf, uint16_t* __restrict__ Cb,
    int ldc, long long sCz,
    int M, int N, int K, int relu)
{
#ifdef HAVE_ASYNC_LDS
    __shared__ uint16_t Ash[2][BM * LSTR];
    __shared__ uint16_t Bsh[2][BN * LSTR];
#else
    __shared__ uint16_t Ash[1][BM * LSTR];
    __shared__ uint16_t Bsh[1][BN * LSTR];
#endif

    const int z = blockIdx.z;
    A  += (long long)z * sAz;
    Bm += (long long)z * sBz;
    const long long coff = (long long)z * sCz;

    const int tm = blockIdx.y * BM, tn = blockIdx.x * BN;
    const int tid  = threadIdx.x;
    const int lane = tid & 31, wid = tid >> 5;
    const int wm = wid >> 2, wn = wid & 3;          // 2x4 wave grid
    const int lrow = tid >> 1, lcol = (tid & 1) * 16;

    f32x8 acc[4][2];
#pragma unroll
    for (int i = 0; i < 4; i++)
#pragma unroll
        for (int j = 0; j < 2; j++)
#pragma unroll
            for (int v = 0; v < 8; v++) acc[i][j][v] = 0.0f;

    const int lh = lane & 15;
    const int ak = (lane < 16) ? 0 : 8;    // A frag: interleaved K per ISA table
    const int bk = (lane < 16) ? 0 : 8;    // B frag: interleaved K per ISA table

    // clamped row addressing: out-of-range rows fetch a valid row; their
    // products land in acc elements that are never stored (store is guarded),
    // and D[m,n] depends only on A row m / B col n, so garbage cannot leak.
    const int arow = (tm + lrow < M) ? (tm + lrow) : (M - 1);
    const int brow = (tn + lrow < N) ? (tn + lrow) : (N - 1);
    const uint16_t* gA = A  + (long long)arow * lda + lcol;
    const uint16_t* gB = Bm + (long long)brow * ldb + lcol;

    auto compute = [&](const uint16_t* AshB, const uint16_t* BshB) {
        union Frag { bf16x16 v; u32x4 q[2]; } af[4], bfr[2];
#pragma unroll
        for (int i = 0; i < 4; i++) {
            const uint16_t* base = &AshB[(wm * 64 + i * 16 + lh) * LSTR];
            af[i].q[0] = *(const u32x4*)(base + ak);        // K 0..7 / 8..15
            af[i].q[1] = *(const u32x4*)(base + ak + 16);   // K 16..23 / 24..31
        }
#pragma unroll
        for (int j = 0; j < 2; j++) {
            const uint16_t* base = &BshB[(wn * 32 + j * 16 + lh) * LSTR];
            bfr[j].q[0] = *(const u32x4*)(base + bk);       // K 0..7 / 8..15
            bfr[j].q[1] = *(const u32x4*)(base + bk + 16);  // K 16..23 / 24..31
        }
#pragma unroll
        for (int i = 0; i < 4; i++)
#pragma unroll
            for (int j = 0; j < 2; j++)
                acc[i][j] = __builtin_amdgcn_wmma_f32_16x16x32_bf16(
                    false, af[i].v, false, bfr[j].v, (short)0, acc[i][j], false, false);
    };

#ifdef HAVE_ASYNC_LDS
    const int nk = K / BK;
    auto issue = [&](int it, int buf) {
        const uint16_t* ga = gA + (long long)it * BK;
        const uint16_t* gb = gB + (long long)it * BK;
        uint16_t* la = &Ash[buf][lrow * LSTR + lcol];
        uint16_t* lb = &Bsh[buf][lrow * LSTR + lcol];
        __builtin_amdgcn_global_load_async_to_lds_b128(to_gbl(ga),     to_lds(la),     0, 0);
        __builtin_amdgcn_global_load_async_to_lds_b128(to_gbl(ga + 8), to_lds(la + 8), 0, 0);
        __builtin_amdgcn_global_load_async_to_lds_b128(to_gbl(gb),     to_lds(lb),     0, 0);
        __builtin_amdgcn_global_load_async_to_lds_b128(to_gbl(gb + 8), to_lds(lb + 8), 0, 0);
    };
    issue(0, 0);
    for (int it = 0; it < nk; ++it) {
        const int cur = it & 1;
        if (it + 1 < nk) {
            issue(it + 1, cur ^ 1);   // target buffer last read two barriers ago
            WAIT_ASYNC(4);            // in-order completion: oldest stage landed
        } else {
            WAIT_ASYNC(0);
        }
        __syncthreads();              // all waves' current-stage LDS writes visible
        compute(&Ash[cur][0], &Bsh[cur][0]);
        __syncthreads();              // all waves done reading before next overwrite
    }
#else
    for (int k0 = 0; k0 < K; k0 += BK) {
        const u32x4* pa = (const u32x4*)(gA + k0);
        u32x4 a0 = pa[0], a1 = pa[1];
        const u32x4* pbm = (const u32x4*)(gB + k0);
        u32x4 b0 = pbm[0], b1 = pbm[1];
        __syncthreads();
        *(u32x4*)&Ash[0][lrow * LSTR + lcol]     = a0;
        *(u32x4*)&Ash[0][lrow * LSTR + lcol + 8] = a1;
        *(u32x4*)&Bsh[0][lrow * LSTR + lcol]     = b0;
        *(u32x4*)&Bsh[0][lrow * LSTR + lcol + 8] = b1;
        __syncthreads();
        compute(&Ash[0][0], &Bsh[0][0]);
    }
#endif

    const int roff = (lane < 16) ? 0 : 8;
#pragma unroll
    for (int j = 0; j < 2; j++) {
        const int col = tn + wn * 32 + j * 16 + lh;
        if (col >= N) continue;
        const float bval = bias ? bias[col] : 0.0f;
#pragma unroll
        for (int i = 0; i < 4; i++) {
            const int rbase = tm + wm * 64 + i * 16 + roff;
#pragma unroll
            for (int v = 0; v < 8; v++) {
                const int row = rbase + v;
                if (row >= M) continue;
                float x = acc[i][j][v] + bval;
                if (relu) x = fmaxf(x, 0.0f);
                const long long idx = coff + (long long)row * ldc + col;
                if (Cf) Cf[idx] = x;
                if (Cb) Cb[idx] = f2bf(x);
            }
        }
    }
}

// ---------------------------------------------------------------------------
// Masked softmax over rows of one batch element's scores [H*S, S] (fp32 in,
// bf16 probs out).  One block per row.
// ---------------------------------------------------------------------------
__global__ __launch_bounds__(256) void k_softmax(
    const float* __restrict__ sc, const uint8_t* __restrict__ mask,
    uint16_t* __restrict__ p, int b, float scale)
{
    const long long row = blockIdx.x;              // h*S + q
    const int q = (int)(row & (S_ - 1));
    const float*   srow = sc + row * (long long)S_;
    const uint8_t* mrow = mask + (long long)b * S_ * S_ + (long long)q * S_;
    uint16_t*      prow = p + row * (long long)S_;

    const int tid = threadIdx.x, lane = tid & 31, wid = tid >> 5;
    __shared__ float red[8];
    float v[8];
    float mx = -3.0e38f;
#pragma unroll
    for (int i = 0; i < 8; i++) {
        const int c = tid + i * 256;
        float s = srow[c] * scale;
        if (mrow[c]) s = -1.0e9f;
        v[i] = s; mx = fmaxf(mx, s);
    }
#pragma unroll
    for (int o = 16; o > 0; o >>= 1) mx = fmaxf(mx, __shfl_xor(mx, o, 32));
    if (lane == 0) red[wid] = mx;
    __syncthreads();
    mx = red[0];
#pragma unroll
    for (int i = 1; i < 8; i++) mx = fmaxf(mx, red[i]);
    __syncthreads();

    float sum = 0.0f;
#pragma unroll
    for (int i = 0; i < 8; i++) { v[i] = __expf(v[i] - mx); sum += v[i]; }
#pragma unroll
    for (int o = 16; o > 0; o >>= 1) sum += __shfl_xor(sum, o, 32);
    if (lane == 0) red[wid] = sum;
    __syncthreads();
    sum = red[0];
#pragma unroll
    for (int i = 1; i < 8; i++) sum += red[i];
    const float inv = 1.0f / sum;
#pragma unroll
    for (int i = 0; i < 8; i++) prow[tid + i * 256] = f2bf(v[i] * inv);
}

// ---------------------------------------------------------------------------
// Fused residual add + LayerNorm: xout = LN(y + resid)*g + beta (fp32 + bf16)
// ---------------------------------------------------------------------------
__global__ __launch_bounds__(256) void k_add_ln(
    const float* __restrict__ y, const float* __restrict__ resid,
    const float* __restrict__ g, const float* __restrict__ beta,
    float* __restrict__ xout, uint16_t* __restrict__ xb)
{
    const long long row = blockIdx.x;
    const float* yr = y + row * D_;
    const float* rr = resid + row * D_;
    float*     xo = xout + row * D_;
    uint16_t* xbo = xb + row * D_;

    const int tid = threadIdx.x, lane = tid & 31, wid = tid >> 5;
    __shared__ float rs[8], rq[8];
    float v[3];
    float s = 0.0f, sq = 0.0f;
#pragma unroll
    for (int i = 0; i < 3; i++) {
        const int c = tid + i * 256;
        const float t = yr[c] + rr[c];
        v[i] = t; s += t; sq += t * t;
    }
#pragma unroll
    for (int o = 16; o > 0; o >>= 1) { s += __shfl_xor(s, o, 32); sq += __shfl_xor(sq, o, 32); }
    if (lane == 0) { rs[wid] = s; rq[wid] = sq; }
    __syncthreads();
    s = 0.0f; sq = 0.0f;
#pragma unroll
    for (int i = 0; i < 8; i++) { s += rs[i]; sq += rq[i]; }
    const float mean = s * (1.0f / D_);
    const float var  = sq * (1.0f / D_) - mean * mean;
    const float rstd = rsqrtf(var + EPS_);
#pragma unroll
    for (int i = 0; i < 3; i++) {
        const int c = tid + i * 256;
        const float o = (v[i] - mean) * rstd * g[c] + beta[c];
        xo[c] = o; xbo[c] = f2bf(o);
    }
}

__global__ __launch_bounds__(256) void k_f2bf(
    const float* __restrict__ in, uint16_t* __restrict__ out, long long n)
{
    const long long i = (long long)blockIdx.x * 256 + threadIdx.x;
    if (i < n) out[i] = f2bf(in[i]);
}

// bf16 V [B,S,D] -> Vt [B,H,HD,S] so ctx GEMM is NT like everything else
__global__ __launch_bounds__(256) void k_vtrans(
    const uint16_t* __restrict__ vb, uint16_t* __restrict__ vt)
{
    const long long i = (long long)blockIdx.x * 256 + threadIdx.x;
    if (i >= (long long)BS_ * D_) return;
    const int d = (int)(i % D_);
    const long long bs = i / D_;
    const int s = (int)(bs % S_);
    const int b = (int)(bs / S_);
    const int h = d / HD_, dd = d % HD_;
    vt[(((long long)b * H_ + h) * HD_ + dd) * S_ + s] = vb[i];
}

// ---------------------------------------------------------------------------
extern "C" void kernel_launch(void* const* d_in, const int* in_sizes, int n_in,
                              void* d_out, int out_size, void* d_ws, size_t ws_size,
                              hipStream_t stream)
{
    (void)in_sizes; (void)n_in; (void)out_size; (void)ws_size;

    const float*   x_in = (const float*)d_in[0];
    const uint8_t* mask = (const uint8_t*)d_in[1];
    const float* Wq = (const float*)d_in[2];  const float* bq = (const float*)d_in[3];
    const float* Wk = (const float*)d_in[4];  const float* bk = (const float*)d_in[5];
    const float* Wv = (const float*)d_in[6];  const float* bv = (const float*)d_in[7];
    const float* Wo = (const float*)d_in[8];  const float* bo = (const float*)d_in[9];
    const float* ln1g = (const float*)d_in[10]; const float* ln1b = (const float*)d_in[11];
    const float* W1 = (const float*)d_in[12]; const float* b1 = (const float*)d_in[13];
    const float* W2 = (const float*)d_in[14]; const float* b2 = (const float*)d_in[15];
    const float* ln2g = (const float*)d_in[16]; const float* ln2b = (const float*)d_in[17];

    char* p = (char*)d_ws;
    auto bump = [&](size_t bytes) -> char* {
        char* r = p; p += (bytes + 255) & ~(size_t)255; return r;
    };
    const long long DD = (long long)D_ * D_;
    const long long FD = (long long)F_ * D_;

    uint16_t* Wqb = (uint16_t*)bump((size_t)L_ * DD * 2);
    uint16_t* Wkb = (uint16_t*)bump((size_t)L_ * DD * 2);
    uint16_t* Wvb = (uint16_t*)bump((size_t)L_ * DD * 2);
    uint16_t* Wob = (uint16_t*)bump((size_t)L_ * DD * 2);
    uint16_t* W1b = (uint16_t*)bump((size_t)L_ * FD * 2);
    uint16_t* W2b = (uint16_t*)bump((size_t)L_ * FD * 2);
    uint16_t* Xb   = (uint16_t*)bump((size_t)BS_ * D_ * 2);
    uint16_t* qb   = (uint16_t*)bump((size_t)BS_ * D_ * 2);
    uint16_t* kb   = (uint16_t*)bump((size_t)BS_ * D_ * 2);
    uint16_t* vb   = (uint16_t*)bump((size_t)BS_ * D_ * 2);
    uint16_t* vt   = (uint16_t*)bump((size_t)BS_ * D_ * 2);
    uint16_t* ctxb = (uint16_t*)bump((size_t)BS_ * D_ * 2);
    uint16_t* hb   = (uint16_t*)bump((size_t)BS_ * F_ * 2);
    float* attnf = (float*)bump((size_t)BS_ * D_ * 4);
    float* fff   = (float*)bump((size_t)BS_ * D_ * 4);
    float* xf    = (float*)bump((size_t)BS_ * D_ * 4);
    float*    sc = (float*)bump((size_t)H_ * S_ * S_ * 4);     // one batch elem
    uint16_t* pb = (uint16_t*)bump((size_t)H_ * S_ * S_ * 2);  // one batch elem

    auto conv = [&](const float* src, uint16_t* dst, long long n) {
        k_f2bf<<<(unsigned)((n + 255) / 256), 256, 0, stream>>>(src, dst, n);
    };
    auto gemm = [&](const uint16_t* A, int lda, long long sAz,
                    const uint16_t* Bmat, int ldb, long long sBz,
                    const float* bias, float* Cf, uint16_t* Cb,
                    int ldc, long long sCz, int M, int N, int K, int relu, int zb) {
        dim3 grid((N + BN - 1) / BN, (M + BM - 1) / BM, zb);
        k_gemm<<<grid, 256, 0, stream>>>(A, lda, sAz, Bmat, ldb, sBz, bias,
                                         Cf, Cb, ldc, sCz, M, N, K, relu);
    };

    // weights -> bf16 once per call
    conv(Wq, Wqb, (long long)L_ * DD);
    conv(Wk, Wkb, (long long)L_ * DD);
    conv(Wv, Wvb, (long long)L_ * DD);
    conv(Wo, Wob, (long long)L_ * DD);
    conv(W1, W1b, (long long)L_ * FD);
    conv(W2, W2b, (long long)L_ * FD);
    conv(x_in, Xb, (long long)BS_ * D_);

    const float scale = 0.125f;  // 1/sqrt(HD)

    for (int i = 0; i < L_; i++) {
        const uint16_t* Wqi = Wqb + i * DD;  const float* bqi = bq + i * D_;
        const uint16_t* Wki = Wkb + i * DD;  const float* bki = bk + i * D_;
        const uint16_t* Wvi = Wvb + i * DD;  const float* bvi = bv + i * D_;
        const uint16_t* Woi = Wob + i * DD;  const float* boi = bo + i * D_;
        const uint16_t* W1i = W1b + i * FD;  const float* b1i = b1 + i * F_;
        const uint16_t* W2i = W2b + i * FD;  const float* b2i = b2 + i * D_;
        const float* resid1 = (i == 0) ? x_in : xf;

        // QKV projections (NT, bf16 out)
        gemm(Xb, D_, 0, Wqi, D_, 0, bqi, nullptr, qb, D_, 0, BS_, D_, D_, 0, 1);
        gemm(Xb, D_, 0, Wki, D_, 0, bki, nullptr, kb, D_, 0, BS_, D_, D_, 0, 1);
        gemm(Xb, D_, 0, Wvi, D_, 0, bvi, nullptr, vb, D_, 0, BS_, D_, D_, 0, 1);
        k_vtrans<<<(unsigned)(((long long)BS_ * D_ + 255) / 256), 256, 0, stream>>>(vb, vt);

        for (int b = 0; b < B_; b++) {
            // scores = q . k^T   (per head via z; fp32 logits)
            gemm(qb + (long long)b * S_ * D_, D_, HD_,
                 kb + (long long)b * S_ * D_, D_, HD_,
                 nullptr, sc, nullptr, S_, (long long)S_ * S_,
                 S_, S_, HD_, 0, H_);
            k_softmax<<<H_ * S_, 256, 0, stream>>>(sc, mask, pb, b, scale);
            // ctx = p . v^T  (vt rows are head dims)
            gemm(pb, S_, (long long)S_ * S_,
                 vt + (long long)b * H_ * HD_ * S_, S_, (long long)HD_ * S_,
                 nullptr, nullptr, ctxb + (long long)b * S_ * D_, D_, HD_,
                 S_, HD_, S_, 0, H_);
        }

        // output projection (fp32 out) + residual LN1
        gemm(ctxb, D_, 0, Woi, D_, 0, boi, attnf, nullptr, D_, 0, BS_, D_, D_, 0, 1);
        k_add_ln<<<BS_, 256, 0, stream>>>(attnf, resid1, ln1g + i * D_, ln1b + i * D_, xf, Xb);

        // FFN
        gemm(Xb, D_, 0, W1i, D_, 0, b1i, nullptr, hb, F_, 0, BS_, F_, D_, 1, 1);
        gemm(hb, F_, 0, W2i, F_, 0, b2i, fff, nullptr, D_, 0, BS_, D_, F_, 0, 1);
        float* xo = (i == L_ - 1) ? (float*)d_out : xf;
        k_add_ln<<<BS_, 256, 0, stream>>>(fff, xf, ln2g + i * D_, ln2b + i * D_, xo, Xb);
    }
}
